// QuantumAttention_25074019074401
// MI455X (gfx1250) — compile-verified, matrix operand order checked by reading, not run on hardware
//
#include <hip/hip_runtime.h>
#include <hip/hip_bf16.h>

// ---------------------------------------------------------------------------
// CDNA5 (gfx1250) wave32 WMMA implementation of "QuantumAttention".
//   q = x @ Wq^T + bq ; k = ctx @ Wk^T + bk ; v = ctx @ Wv^T + bv
//   e = tanh(cat(q,k) @ We^T + be)
//   out = softmax(e @ v^T / sqrt(D)) @ v
// bf16 data path through v_wmma_f32_16x16x32_bf16, fp32 accumulation.
// Tile staging uses gfx1250 async global->LDS DMA (ASYNCcnt) when available.
// ---------------------------------------------------------------------------

typedef __attribute__((ext_vector_type(16))) __bf16 v16bf;
typedef __attribute__((ext_vector_type(8)))  float  v8f;

#define WMMA_BF16(A, B, C) \
  __builtin_amdgcn_wmma_f32_16x16x32_bf16(false, (A), false, (B), (short)0, (C), false, false)

// --- async global->LDS staging (gfx1250 ASYNCcnt path) ---------------------
#if defined(__has_builtin)
#if __has_builtin(__builtin_amdgcn_global_load_async_to_lds_b128) && \
    __has_builtin(__builtin_amdgcn_s_wait_asynccnt)
#define HAVE_ASYNC_LDS 1
#endif
#endif

#if defined(HAVE_ASYNC_LDS)
typedef int v4i __attribute__((vector_size(16)));
typedef __attribute__((address_space(1))) v4i* as1_v4i_ptr;  // global
typedef __attribute__((address_space(3))) v4i* as3_v4i_ptr;  // LDS
#endif

__device__ inline void copy16_async(const __bf16* gsrc, __bf16* ldst) {
#if defined(HAVE_ASYNC_LDS)
  // Flat shared addresses carry the LDS offset in the low 32 bits.
  __builtin_amdgcn_global_load_async_to_lds_b128(
      (as1_v4i_ptr)(uintptr_t)gsrc,
      (as3_v4i_ptr)(unsigned int)(uintptr_t)ldst, 0, 0);
#else
  *(uint4*)ldst = *(const uint4*)gsrc;
#endif
}
__device__ inline void async_join() {
#if defined(HAVE_ASYNC_LDS)
  __builtin_amdgcn_s_wait_asynccnt(0);
#endif
}

// --- fragment loaders (ISA 7.12.2 layouts, wave32) -------------------------
// A-matrix 16x32 bf16 (also B when the source is stored [N][K] row-major):
// lane L: row = L&15, half = L>>4 selects K-halves; VGPR v holds K pairs.
__device__ inline v16bf frag_nk(const __bf16* base, int stride, int lane) {
  const int r = lane & 15, h = lane >> 4;
  union { v16bf v; unsigned int u[8]; } f;
#pragma unroll
  for (int v = 0; v < 8; ++v) {
    const int k = ((v & 4) ? 16 : 0) + h * 8 + 2 * (v & 3);
    f.u[v] = *(const unsigned int*)&base[r * stride + k];  // 2 bf16, 4B aligned
  }
  return f.v;
}

// B operand when source is stored [K][N] row-major (col gather).
__device__ inline v16bf frag_kn(const __bf16* base, int stride, int lane) {
  const int n = lane & 15, h = lane >> 4;
  v16bf f;
#pragma unroll
  for (int v = 0; v < 8; ++v) {
    const int k = ((v & 4) ? 16 : 0) + h * 8 + 2 * (v & 3);
    f[2 * v]     = base[k * stride + n];
    f[2 * v + 1] = base[(k + 1) * stride + n];
  }
  return f;
}

// 16x16 f32 C/D tile: VGPR j, lane L -> row = j + 8*(L>>4), col = L&15.
__device__ inline v8f acc_load(const float* base, int stride, int lane) {
  const int c = lane & 15, h = lane >> 4;
  v8f a;
#pragma unroll
  for (int j = 0; j < 8; ++j) a[j] = base[(j + 8 * h) * stride + c];
  return a;
}
__device__ inline void acc_store(float* base, int stride, int lane, v8f a) {
  const int c = lane & 15, h = lane >> 4;
#pragma unroll
  for (int j = 0; j < 8; ++j) base[(j + 8 * h) * stride + c] = a[j];
}

// --- fp32 -> bf16 convert ---------------------------------------------------
__global__ void cvt_bf16_kernel(const float* __restrict__ in,
                                __bf16* __restrict__ out, int n) {
  int i = blockIdx.x * blockDim.x + threadIdx.x;
  if (i < n) out[i] = (__bf16)in[i];
}

// --- generic NT GEMM: C[m, coff+n] = act(A[M,K] @ W[N,K]^T + bias[n]) ------
// 64x128 block tile, 8 waves; each wave owns four independent 16x16 WMMA
// tiles (no WMMA->WMMA accumulator hazard between adjacent instructions).
template <bool TANH>
__global__ void gemm_nt_kernel(const __bf16* __restrict__ A, int lda,
                               const __bf16* __restrict__ W, int ldw,
                               const float* __restrict__ bias,
                               __bf16* __restrict__ C, int ldc, int coff,
                               int K) {
  __shared__ __align__(16) __bf16 As[64 * 40];
  __shared__ __align__(16) __bf16 Ws[128 * 40];
  const int tid = threadIdx.x, lane = tid & 31, w = tid >> 5;
  const int m0 = blockIdx.y * 64, n0 = blockIdx.x * 128;
  const int wm = w & 3, wn0 = (w >> 2) * 4;
  const int arow = tid >> 2, acol = (tid & 3) * 8;  // A: 256 chunks of 8

  v8f acc[4] = {};
  for (int k0 = 0; k0 < K; k0 += 32) {
    copy16_async(&A[(size_t)(m0 + arow) * lda + k0 + acol], &As[arow * 40 + acol]);
#pragma unroll
    for (int j = 0; j < 2; ++j) {                   // W: 512 chunks of 8
      const int cch = tid * 2 + j, wrow = cch >> 2, wcol = (cch & 3) * 8;
      copy16_async(&W[(size_t)(n0 + wrow) * ldw + k0 + wcol], &Ws[wrow * 40 + wcol]);
    }
    async_join();
    __syncthreads();
    v16bf a = frag_nk(&As[wm * 16 * 40], 40, lane);
    v16bf b[4];
#pragma unroll
    for (int t = 0; t < 4; ++t)
      b[t] = frag_nk(&Ws[((wn0 + t) * 16) * 40], 40, lane);
#pragma unroll
    for (int t = 0; t < 4; ++t)
      acc[t] = WMMA_BF16(a, b[t], acc[t]);
    __syncthreads();
  }
  const int h = lane >> 4, c = lane & 15;
#pragma unroll
  for (int t = 0; t < 4; ++t) {
#pragma unroll
    for (int j = 0; j < 8; ++j) {
      const int m = m0 + wm * 16 + j + 8 * h;
      const int n = n0 + (wn0 + t) * 16 + c;
      float v = acc[t][j] + bias[n];
      if (TANH) v = tanhf(v);
      C[(size_t)m * ldc + coff + n] = (__bf16)v;
    }
  }
}

// --- flash attention: out = softmax(E @ V^T * scale) @ V -------------------
// One block = one batch x 16 query rows. kv consumed in chunks of 32 so the
// P@V update is an exact K=32 WMMA. ~185 KB dynamic LDS (<320 KB/WGP).
__global__ void flash_attn_kernel(const __bf16* __restrict__ E,
                                  const __bf16* __restrict__ V,
                                  float* __restrict__ Out) {
  constexpr int S = 2048, D = 1024;
  constexpr int ELD = 1032, VLD = 1032, ALD = 1040, PLD = 40;
  extern __shared__ char smem[];
  __bf16* e_s = (__bf16*)smem;                 // 16 x ELD
  __bf16* v_s = e_s + 16 * ELD;                // 32 x VLD
  float*  acc = (float*)(v_s + 32 * VLD);      // 16 x ALD
  float*  sp  = acc + 16 * ALD;                // 8 x 16 x 32 partial scores
  float*  sS  = sp + 8 * 512;                  // 16 x 32 reduced scores
  float*  mrow = sS + 512;                     // 16
  float*  lrow = mrow + 16;                    // 16
  float*  arow = lrow + 16;                    // 16
  __bf16* p_s = (__bf16*)(arow + 16);          // 16 x PLD

  const int tid = threadIdx.x, lane = tid & 31, w = tid >> 5;
  const int b = blockIdx.y, q0 = blockIdx.x * 16;
  const size_t baseE = ((size_t)b * S + q0) * D;
  const size_t baseV = (size_t)b * S * D;

  for (int i = tid; i < 16 * (D / 8); i += 256) {
    const int r = i >> 7, c8 = (i & 127) * 8;
    copy16_async(&E[baseE + (size_t)r * D + c8], &e_s[r * ELD + c8]);
  }
  for (int i = tid; i < 16 * D; i += 256)
    acc[(i >> 10) * ALD + (i & 1023)] = 0.f;
  if (tid < 16) { mrow[tid] = -3.0e38f; lrow[tid] = 0.f; }
  async_join();
  __syncthreads();

  // per-wave score A-fragments (K range [w*128, w*128+128)), kv-loop invariant
  v16bf fa[4];
#pragma unroll
  for (int ks = 0; ks < 4; ++ks)
    fa[ks] = frag_nk(&e_s[w * 128 + ks * 32], ELD, lane);

  const float scale = 0.03125f;  // 1/sqrt(1024)

  for (int jt = 0; jt < S / 32; ++jt) {
    const int kv0 = jt * 32;
    __syncthreads();
    for (int i = tid; i < 32 * (D / 8); i += 256) {       // stage V chunk
      const int r = i >> 7, c8 = (i & 127) * 8;
      copy16_async(&V[baseV + (size_t)(kv0 + r) * D + c8], &v_s[r * VLD + c8]);
    }
    async_join();
    __syncthreads();
    // partial scores: wave w reduces its K slice; the two 16-col tiles are
    // interleaved so adjacent WMMAs use independent accumulators (no hazard).
    {
      v8f s0 = {}; v8f s1 = {};
#pragma unroll
      for (int ks = 0; ks < 4; ++ks) {
        v16bf fb0 = frag_nk(&v_s[w * 128 + ks * 32], VLD, lane);
        v16bf fb1 = frag_nk(&v_s[16 * VLD + w * 128 + ks * 32], VLD, lane);
        s0 = WMMA_BF16(fa[ks], fb0, s0);
        s1 = WMMA_BF16(fa[ks], fb1, s1);
      }
      acc_store(&sp[w * 512 + 0], 32, lane, s0);
      acc_store(&sp[w * 512 + 16], 32, lane, s1);
    }
    __syncthreads();
    for (int i = tid; i < 512; i += 256) {                // cross-wave reduce
      float v = 0.f;
#pragma unroll
      for (int ww = 0; ww < 8; ++ww) v += sp[ww * 512 + i];
      sS[i] = v * scale;
    }
    __syncthreads();
    if (tid < 16) {                                       // online softmax row
      const int r = tid;
      float mx = mrow[r];
#pragma unroll
      for (int c = 0; c < 32; ++c) mx = fmaxf(mx, sS[r * 32 + c]);
      const float al = __expf(mrow[r] - mx);
      float ps = 0.f;
#pragma unroll
      for (int c = 0; c < 32; ++c) {
        const float p = __expf(sS[r * 32 + c] - mx);
        p_s[r * PLD + c] = (__bf16)p;
        ps += p;
      }
      lrow[r] = lrow[r] * al + ps;
      mrow[r] = mx;
      arow[r] = al;
    }
    __syncthreads();
    for (int i = tid; i < 16 * D; i += 256) {             // rescale accum
      const int r = i >> 10, c = i & 1023;
      acc[r * ALD + c] *= arow[r];
    }
    __syncthreads();
    // acc[16, w*128 : w*128+128] += P(16x32) @ Vchunk(32x128)
    v16bf fp = frag_nk(p_s, PLD, lane);
#pragma unroll
    for (int nt = 0; nt < 8; ++nt) {
      const int nc = w * 128 + nt * 16;
      v8f cac = acc_load(&acc[nc], ALD, lane);
      v16bf fb = frag_kn(&v_s[nc], VLD, lane);
      cac = WMMA_BF16(fp, fb, cac);
      acc_store(&acc[nc], ALD, lane, cac);
    }
  }
  __syncthreads();
  for (int i = tid; i < 16 * D; i += 256) {
    const int r = i >> 10, c = i & 1023;
    Out[((size_t)b * S + q0 + r) * D + c] = acc[r * ALD + c] / lrow[r];
  }
}

// ---------------------------------------------------------------------------
extern "C" void kernel_launch(void* const* d_in, const int* in_sizes, int n_in,
                              void* d_out, int out_size, void* d_ws, size_t ws_size,
                              hipStream_t stream) {
  const float* x   = (const float*)d_in[0];
  const float* ctx = (const float*)d_in[1];
  const float* Wq  = (const float*)d_in[2];
  const float* bq  = (const float*)d_in[3];
  const float* Wk  = (const float*)d_in[4];
  const float* bk  = (const float*)d_in[5];
  const float* Wv  = (const float*)d_in[6];
  const float* bv  = (const float*)d_in[7];
  const float* We  = (const float*)d_in[8];
  const float* be  = (const float*)d_in[9];
  float* out = (float*)d_out;

  const int B = 8, S = 2048, D = 1024;
  const size_t M = (size_t)B * S;  // 16384

  char* ws = (char*)d_ws;
  auto carve = [&](size_t bytes) {
    char* p = ws;
    ws += (bytes + 255) & ~(size_t)255;
    return p;
  };
  __bf16* xb  = (__bf16*)carve(M * D * 2);           // x bf16
  __bf16* cb  = (__bf16*)carve(M * D * 2);           // context bf16
  __bf16* wqb = (__bf16*)carve((size_t)D * D * 2);
  __bf16* wkb = (__bf16*)carve((size_t)D * D * 2);
  __bf16* wvb = (__bf16*)carve((size_t)D * D * 2);
  __bf16* web = (__bf16*)carve((size_t)D * 2 * D * 2);
  __bf16* qk  = (__bf16*)carve(M * 2 * D * 2);       // [M, 2D]: q | k
  __bf16* vb  = (__bf16*)carve(M * D * 2);
  __bf16* eb  = (__bf16*)carve(M * D * 2);

  auto cvt = [&](const float* src, __bf16* dst, int n) {
    cvt_bf16_kernel<<<(n + 255) / 256, 256, 0, stream>>>(src, dst, n);
  };
  cvt(x,   xb,  (int)(M * D));
  cvt(ctx, cb,  (int)(M * D));
  cvt(Wq,  wqb, D * D);
  cvt(Wk,  wkb, D * D);
  cvt(Wv,  wvb, D * D);
  cvt(We,  web, D * 2 * D);

  dim3 g(D / 128, (unsigned)(M / 64));  // (8, 256)
  gemm_nt_kernel<false><<<g, 256, 0, stream>>>(xb, D,     wqb, D,     bq, qk, 2 * D, 0, D);
  gemm_nt_kernel<false><<<g, 256, 0, stream>>>(cb, D,     wkb, D,     bk, qk, 2 * D, D, D);
  gemm_nt_kernel<false><<<g, 256, 0, stream>>>(cb, D,     wvb, D,     bv, vb, D,     0, D);
  gemm_nt_kernel<true ><<<g, 256, 0, stream>>>(qk, 2 * D, web, 2 * D, be, eb, D,     0, 2 * D);

  const size_t smem = (16 * 1032 + 32 * 1032) * sizeof(__bf16) +   // e_s + v_s
                      (16 * 1040 + 8 * 512 + 512 + 48) * sizeof(float) +
                      16 * 40 * sizeof(__bf16);                    // p tile
  (void)hipFuncSetAttribute((const void*)flash_attn_kernel,
                            hipFuncAttributeMaxDynamicSharedMemorySize, (int)smem);
  flash_attn_kernel<<<dim3(S / 16, B), 256, smem, stream>>>(eb, vb, out);
}
